// BoxHead_50483045597750
// MI455X (gfx1250) — compile-verified, matrix operand order checked by reading, not run on hardware
//
#include <hip/hip_runtime.h>
#include <hip/hip_bf16.h>
#include <math.h>

typedef __attribute__((ext_vector_type(16))) _Float16 v16h;
typedef __attribute__((ext_vector_type(8)))  _Float16 v8h;
typedef __attribute__((ext_vector_type(8)))  float    v8f;

#define ROIS        2000
#define CH          256
#define POS         49
#define MPAD        64
#define ROI_STRIDE  (MPAD * CH)      // 16384 halfs per roi
#define FC1_K       (POS * CH)       // 12544
#define FC_N        1024
#define LSTR        264              // padded LDS row stride (halfs), 16B aligned

union U16h { v16h v; v8h h[2]; };

// Load one lane's 16-half WMMA fragment. `base` points at the k-block start of
// this lane's row (A) or column (B); hi = lane>=16 selects the K-half per the
// CDNA5 16-bit A/B layout (VGPR0-3 = K[hi*8 .. hi*8+7], VGPR4-7 = +16).
__device__ __forceinline__ v16h load_frag(const _Float16* __restrict__ base, int hi) {
    U16h u;
    const _Float16* p = base + hi * 8;
    u.h[0] = *(const v8h*)(p);
    u.h[1] = *(const v8h*)(p + 16);
    return u.v;
}

// ---------------- weight conversion / repack ----------------
__global__ void cvt_plain(const float* __restrict__ s, _Float16* __restrict__ d, int n) {
    int i = blockIdx.x * blockDim.x + threadIdx.x;
    if (i < n) d[i] = (_Float16)s[i];
}

// w[o][i][ky][kx] (256,256,3,3) -> d[o][tap*256 + i], tap = ky*3+kx
__global__ void cvt_conv3(const float* __restrict__ s, _Float16* __restrict__ d) {
    int i = blockIdx.x * blockDim.x + threadIdx.x;
    if (i >= 256 * 2304) return;
    int o = i / 2304, r = i % 2304, tap = r >> 8, c = r & 255;
    d[i] = (_Float16)s[(o * 256 + c) * 9 + tap];
}

// wf1[f][c*49 + p] -> d[f][p*256 + c]
__global__ void cvt_wf1(const float* __restrict__ s, _Float16* __restrict__ d) {
    int i = blockIdx.x * blockDim.x + threadIdx.x;
    if (i >= 1024 * 12544) return;
    int f = i / 12544, r = i % 12544, p = r >> 8, c = r & 255;
    d[i] = (_Float16)s[f * 12544 + c * 49 + p];
}

// ---------------- ROI align -> f16 [roi][pos(64)][ch] ----------------
__global__ __launch_bounds__(256) void roi_align_k(
        const float* __restrict__ feat, const float* __restrict__ props,
        _Float16* __restrict__ X0) {
    int roi = blockIdx.x;
    int c   = threadIdx.x;
    int b   = roi / 1000;
    const float* pr = props + (size_t)roi * 4;
    float bx = pr[0], by = pr[1], bw = pr[2], bh = pr[3];
    const float SC = 1.0f / 32.0f;
    float rx1 = bx * SC - 0.5f,        ry1 = by * SC - 0.5f;
    float rx2 = (bx + bw) * SC - 0.5f, ry2 = (by + bh) * SC - 0.5f;
    float cellh = (ry2 - ry1) / 7.0f, cellw = (rx2 - rx1) / 7.0f;
    const float* fb = feat + ((size_t)b * 256 + c) * 1024;
    _Float16* drow = X0 + (size_t)roi * ROI_STRIDE;

    for (int oy = 0; oy < 7; ++oy)
    for (int ox = 0; ox < 7; ++ox) {
        float sum = 0.0f;
#pragma unroll
        for (int sy = 0; sy < 2; ++sy)
#pragma unroll
        for (int sx = 0; sx < 2; ++sx) {
            float ys = ry1 + ((float)(oy * 2 + sy) + 0.5f) * cellh * 0.5f;
            float xs = rx1 + ((float)(ox * 2 + sx) + 0.5f) * cellw * 0.5f;
            bool vy = (ys > -1.0f) && (ys < 32.0f);
            bool vx = (xs > -1.0f) && (xs < 32.0f);
            float fy = fminf(fmaxf(ys, 0.0f), 31.0f);
            float fx = fminf(fmaxf(xs, 0.0f), 31.0f);
            int iy0 = (int)floorf(fy), ix0 = (int)floorf(fx);
            int iy1 = min(iy0 + 1, 31), ix1 = min(ix0 + 1, 31);
            float ly = fy - (float)iy0, lx = fx - (float)ix0;
            float hy = 1.0f - ly,       hx = 1.0f - lx;
            float v = fb[iy0 * 32 + ix0] * hy * hx + fb[iy0 * 32 + ix1] * hy * lx +
                      fb[iy1 * 32 + ix0] * ly * hx + fb[iy1 * 32 + ix1] * ly * lx;
            sum += (vy && vx) ? v : 0.0f;
        }
        drow[(oy * 7 + ox) * CH + c] = (_Float16)(sum * 0.25f);
    }
    for (int p = POS; p < MPAD; ++p) drow[p * CH + c] = (_Float16)0.0f;
}

// ---------------- conv (1x1 or 3x3) as implicit-GEMM WMMA ----------------
// One workgroup per ROI. Input tile staged in LDS with an extra all-zero row
// (index MPAD) so out-of-bounds im2col taps become a pointer select, not a
// divergent branch. 8 waves x 2 N-tiles; each wave holds 4 M-tile
// accumulators so every B fragment is reused 4x. All 4 A fragments are
// loaded into distinct registers before the wmma group so all ds_loads are
// in flight simultaneously (graduated dscnt waits instead of dscnt==0).
__global__ __launch_bounds__(256) void conv_gemm(
        const _Float16* __restrict__ Xin, const _Float16* __restrict__ Wc,
        const float* __restrict__ bias, _Float16* __restrict__ Xout, int ntaps) {
    __shared__ _Float16 Xl[(MPAD + 1) * LSTR];   // 34.3 KB (row MPAD = zeros)
    int roi = blockIdx.x;
    const _Float16* src = Xin + (size_t)roi * ROI_STRIDE;
    for (int chunk = threadIdx.x; chunk < (MPAD + 1) * 32; chunk += 256) {
        int p = chunk >> 5, off = (chunk & 31) * 8;
        v8h val;
        if (p < MPAD) {
            val = *(const v8h*)(src + p * CH + off);
        } else {
#pragma unroll
            for (int i = 0; i < 8; ++i) val[i] = (_Float16)0.0f;
        }
        *(v8h*)&Xl[p * LSTR + off] = val;
    }
    __syncthreads();

    int wave = threadIdx.x >> 5, lane = threadIdx.x & 31;
    int l15 = lane & 15, hi = lane >> 4;
    int K = ntaps * 256;
    _Float16* dst = Xout + (size_t)roi * ROI_STRIDE;

    // per-M-tile lane row coordinates (computed once)
    int py_[4], px_[4];
#pragma unroll
    for (int mt = 0; mt < 4; ++mt) {
        int p = mt * 16 + l15;
        py_[mt] = p / 7;
        px_[mt] = p % 7;
    }

    for (int nt = wave; nt < 16; nt += 8) {
        v8f acc[4];
#pragma unroll
        for (int mt = 0; mt < 4; ++mt)
#pragma unroll
            for (int i = 0; i < 8; ++i) acc[mt][i] = 0.0f;

        int col = nt * 16 + l15;
        const _Float16* wrow = Wc + (size_t)col * K;

        for (int tap = 0; tap < ntaps; ++tap) {
            int dy = (ntaps == 9) ? (tap / 3 - 1) : 0;
            int dx = (ntaps == 9) ? (tap - (tap / 3) * 3 - 1) : 0;
            const _Float16* ap[4];
#pragma unroll
            for (int mt = 0; mt < 4; ++mt) {
                int yy = py_[mt] + dy, xx = px_[mt] + dx;
                bool valid = (ntaps == 1) ||
                             (yy >= 0 && yy < 7 && xx >= 0 && xx < 7);
                int p2 = (ntaps == 1) ? (mt * 16 + l15)
                                      : (valid ? (yy * 7 + xx) : MPAD);
                ap[mt] = &Xl[p2 * LSTR];
            }
            const _Float16* wtap = wrow + tap * 256;
#pragma unroll
            for (int kk = 0; kk < 8; ++kk) {
                int cb = kk * 32;
                // issue all loads first -> distinct regs, overlapped waits
                v16h b  = load_frag(wtap + cb, hi);
                v16h a0 = load_frag(ap[0] + cb, hi);
                v16h a1 = load_frag(ap[1] + cb, hi);
                v16h a2 = load_frag(ap[2] + cb, hi);
                v16h a3 = load_frag(ap[3] + cb, hi);
                acc[0] = __builtin_amdgcn_wmma_f32_16x16x32_f16(
                    false, a0, false, b, (short)0, acc[0], false, false);
                acc[1] = __builtin_amdgcn_wmma_f32_16x16x32_f16(
                    false, a1, false, b, (short)0, acc[1], false, false);
                acc[2] = __builtin_amdgcn_wmma_f32_16x16x32_f16(
                    false, a2, false, b, (short)0, acc[2], false, false);
                acc[3] = __builtin_amdgcn_wmma_f32_16x16x32_f16(
                    false, a3, false, b, (short)0, acc[3], false, false);
            }
        }
        float bv = bias[col];
#pragma unroll
        for (int mt = 0; mt < 4; ++mt)
#pragma unroll
            for (int v = 0; v < 8; ++v) {
                int row = mt * 16 + hi * 8 + v;
                float val = acc[mt][v] + bv;
                val = val > 0.0f ? val : 0.0f;
                dst[row * CH + col] = (_Float16)val;
            }
    }
}

// ---------------- FC GEMM: M=2000 (125 tiles), N=1024 (64 tiles) ----------
// One wave per (M-tile, group of 4 N-tiles): A fragment loaded once per
// k-step and reused across 4 v_wmma. 250 blocks x 8 waves = 2000 jobs.
__global__ __launch_bounds__(256) void fc_gemm(
        const _Float16* __restrict__ A, int strideA,
        const _Float16* __restrict__ W, int K,
        const float* __restrict__ bias, _Float16* __restrict__ Out, int relu) {
    int wave = threadIdx.x >> 5, lane = threadIdx.x & 31;
    int l15 = lane & 15, hi = lane >> 4;
    int tile = blockIdx.x * 8 + wave;
    int mt = tile >> 4;          // 0..124
    int ng = tile & 15;          // 0..15 -> N-tiles ng*4 .. ng*4+3
    int row0 = mt * 16;
    const _Float16* arow = A + (size_t)(row0 + l15) * strideA;

    int col[4];
    const _Float16* wrow[4];
#pragma unroll
    for (int j = 0; j < 4; ++j) {
        col[j]  = (ng * 4 + j) * 16 + l15;
        wrow[j] = W + (size_t)col[j] * K;
    }

    v8f acc[4];
#pragma unroll
    for (int j = 0; j < 4; ++j)
#pragma unroll
        for (int i = 0; i < 8; ++i) acc[j][i] = 0.0f;

    for (int kb = 0; kb < K; kb += 32) {
        v16h a  = load_frag(arow + kb, hi);
        v16h b0 = load_frag(wrow[0] + kb, hi);
        v16h b1 = load_frag(wrow[1] + kb, hi);
        v16h b2 = load_frag(wrow[2] + kb, hi);
        v16h b3 = load_frag(wrow[3] + kb, hi);
        acc[0] = __builtin_amdgcn_wmma_f32_16x16x32_f16(
            false, a, false, b0, (short)0, acc[0], false, false);
        acc[1] = __builtin_amdgcn_wmma_f32_16x16x32_f16(
            false, a, false, b1, (short)0, acc[1], false, false);
        acc[2] = __builtin_amdgcn_wmma_f32_16x16x32_f16(
            false, a, false, b2, (short)0, acc[2], false, false);
        acc[3] = __builtin_amdgcn_wmma_f32_16x16x32_f16(
            false, a, false, b3, (short)0, acc[3], false, false);
    }
#pragma unroll
    for (int j = 0; j < 4; ++j) {
        float bv = bias[col[j]];
#pragma unroll
        for (int v = 0; v < 8; ++v) {
            int row = row0 + hi * 8 + v;
            float val = acc[j][v] + bv;
            if (relu) val = val > 0.0f ? val : 0.0f;
            Out[(size_t)row * FC_N + col[j]] = (_Float16)val;
        }
    }
}

// ---------------- heads + box regression + clip ----------------
__global__ __launch_bounds__(256) void heads_k(
        const _Float16* __restrict__ F2,
        const float* __restrict__ wobj, const float* __restrict__ bobj,
        const float* __restrict__ wreg, const float* __restrict__ breg,
        const float* __restrict__ props, float* __restrict__ out) {
    int roi = blockIdx.x * blockDim.x + threadIdx.x;
    if (roi >= ROIS) return;
    const _Float16* f = F2 + (size_t)roi * 1024;
    float so = bobj[0];
    float r0 = breg[0], r1 = breg[1], r2 = breg[2], r3 = breg[3];
    for (int k = 0; k < 1024; ++k) {
        float v = (float)f[k];
        so += v * wobj[k];
        r0 += v * wreg[k];
        r1 += v * wreg[1024 + k];
        r2 += v * wreg[2048 + k];
        r3 += v * wreg[3072 + k];
    }
    float obj = 1.0f / (1.0f + __expf(-so));
    const float* pr = props + (size_t)roi * 4;
    float px = pr[0], py = pr[1], pw = pr[2], ph = pr[3];
    float d0 = r0 * 0.1f, d1 = r1 * 0.1f, d2 = r2 * 0.2f, d3 = r3 * 0.2f;
    float cx = px + 0.5f * pw, cy = py + 0.5f * ph;
    float nw = pw * __expf(d2), nh = ph * __expf(d3);
    float ncx = d0 * pw + cx,   ncy = d1 * ph + cy;
    float ox = ncx - 0.5f * nw, oy = ncy - 0.5f * nh;
    float cxl = fminf(fmaxf(ox, 0.0f), 1024.0f);
    float cyl = fminf(fmaxf(oy, 0.0f), 1024.0f);
    float cwl = fminf(fmaxf(nw, 0.0f), 1024.0f - cxl);
    float chl = fminf(fmaxf(nh, 0.0f), 1024.0f - cyl);
    out[(size_t)roi * 4 + 0] = cxl;
    out[(size_t)roi * 4 + 1] = cyl;
    out[(size_t)roi * 4 + 2] = cwl;
    out[(size_t)roi * 4 + 3] = chl;
    out[8000 + roi] = obj;
}

// ---------------- NMS per image: bitonic sort + greedy suppress ----------
__global__ __launch_bounds__(1024) void nms_k(float* __restrict__ out) {
    int b = blockIdx.x, t = threadIdx.x;
    __shared__ float ss[1024];
    __shared__ int   id[1024];
    __shared__ float BX1[1024], BY1[1024], BX2[1024], BY2[1024], AR[1024];
    __shared__ int   keep[1024];

    ss[t] = (t < 1000) ? out[8000 + b * 1000 + t] : -1e30f;
    id[t] = t;
    __syncthreads();

    for (int k = 2; k <= 1024; k <<= 1) {
        for (int j = k >> 1; j > 0; j >>= 1) {
            int ixj = t ^ j;
            if (ixj > t) {
                bool desc = ((t & k) == 0);
                float a = ss[t], c = ss[ixj];
                bool sw = desc ? (a < c) : (a > c);
                if (sw) {
                    ss[t] = c; ss[ixj] = a;
                    int tmp = id[t]; id[t] = id[ixj]; id[ixj] = tmp;
                }
            }
            __syncthreads();
        }
    }

    if (t < 1000) {
        int j = id[t];
        const float* bb = out + (size_t)(b * 1000 + j) * 4;
        float x1 = bb[0], y1v = bb[1], w = bb[2], h = bb[3];
        BX1[t] = x1; BY1[t] = y1v; BX2[t] = x1 + w; BY2[t] = y1v + h;
        AR[t]  = fmaxf(w, 0.0f) * fmaxf(h, 0.0f);
        keep[t] = 1;
    } else {
        BX1[t] = 0; BY1[t] = 0; BX2[t] = 0; BY2[t] = 0; AR[t] = 0; keep[t] = 0;
    }
    __syncthreads();

    for (int i = 0; i < 999; ++i) {
        if (t > i && t < 1000 && keep[t] && keep[i]) {
            float lx = fmaxf(BX1[t], BX1[i]);
            float ly = fmaxf(BY1[t], BY1[i]);
            float rx = fminf(BX2[t], BX2[i]);
            float ry = fminf(BY2[t], BY2[i]);
            float iw = fmaxf(rx - lx, 0.0f), ih = fmaxf(ry - ly, 0.0f);
            float inter = iw * ih;
            float iou = inter / fmaxf(AR[t] + AR[i] - inter, 1e-9f);
            if (iou > 0.5f) keep[t] = 0;
        }
        __syncthreads();
    }
    if (t < 1000)
        out[10000 + b * 1000 + id[t]] = (keep[t] && (ss[t] > 0.05f)) ? 1.0f : 0.0f;
}

// ---------------- host launcher ----------------
extern "C" void kernel_launch(void* const* d_in, const int* in_sizes, int n_in,
                              void* d_out, int out_size, void* d_ws, size_t ws_size,
                              hipStream_t stream) {
    const float* feat  = (const float*)d_in[0];
    const float* props = (const float*)d_in[1];
    const float* w11   = (const float*)d_in[2];
    const float* b11   = (const float*)d_in[3];
    const float* w1    = (const float*)d_in[4];
    const float* b1    = (const float*)d_in[5];
    const float* w2    = (const float*)d_in[6];
    const float* b2    = (const float*)d_in[7];
    const float* wf1   = (const float*)d_in[8];
    const float* bf1   = (const float*)d_in[9];
    const float* wf2   = (const float*)d_in[10];
    const float* bf2   = (const float*)d_in[11];
    const float* wobj  = (const float*)d_in[12];
    const float* bobj  = (const float*)d_in[13];
    const float* wreg  = (const float*)d_in[14];
    const float* breg  = (const float*)d_in[15];
    float* out = (float*)d_out;

    char* ws = (char*)d_ws;
    size_t off = 0;
    auto take = [&](size_t bytes) -> char* {
        char* p = ws + off;
        off += (bytes + 255) & ~(size_t)255;
        return p;
    };
    _Float16* W11c = (_Float16*)take((size_t)256 * 256 * 2);
    _Float16* W1c  = (_Float16*)take((size_t)256 * 2304 * 2);
    _Float16* W2c  = (_Float16*)take((size_t)256 * 2304 * 2);
    _Float16* WF1c = (_Float16*)take((size_t)1024 * 12544 * 2);
    _Float16* WF2c = (_Float16*)take((size_t)1024 * 1024 * 2);
    _Float16* X0   = (_Float16*)take((size_t)ROIS * ROI_STRIDE * 2);
    _Float16* X1   = (_Float16*)take((size_t)ROIS * ROI_STRIDE * 2);
    _Float16* F1   = (_Float16*)take((size_t)ROIS * 1024 * 2);
    _Float16* F2   = (_Float16*)take((size_t)ROIS * 1024 * 2);

    cvt_plain<<<(65536 + 255) / 256, 256, 0, stream>>>(w11, W11c, 65536);
    cvt_conv3<<<(589824 + 255) / 256, 256, 0, stream>>>(w1, W1c);
    cvt_conv3<<<(589824 + 255) / 256, 256, 0, stream>>>(w2, W2c);
    cvt_wf1<<<(12845056 + 255) / 256, 256, 0, stream>>>(wf1, WF1c);
    cvt_plain<<<(1048576 + 255) / 256, 256, 0, stream>>>(wf2, WF2c, 1048576);

    roi_align_k<<<ROIS, 256, 0, stream>>>(feat, props, X0);

    conv_gemm<<<ROIS, 256, 0, stream>>>(X0, W11c, b11, X1, 1);   // 1x1
    conv_gemm<<<ROIS, 256, 0, stream>>>(X1, W1c,  b1,  X0, 9);   // 3x3
    conv_gemm<<<ROIS, 256, 0, stream>>>(X0, W2c,  b2,  X1, 9);   // 3x3

    fc_gemm<<<250, 256, 0, stream>>>(X1, ROI_STRIDE, WF1c, FC1_K, bf1, F1, 1);
    fc_gemm<<<250, 256, 0, stream>>>(F1, 1024,       WF2c, 1024,  bf2, F2, 1);

    heads_k<<<(ROIS + 255) / 256, 256, 0, stream>>>(F2, wobj, bobj, wreg, breg, props, out);
    nms_k<<<2, 1024, 0, stream>>>(out);
}
